// M2BertModel_81389630259255
// MI455X (gfx1250) — matrix operand order repaired and checked by reading, not compiled, not run on hardware
//
#include <hip/hip_runtime.h>
#include <hip/hip_bf16.h>
#include <math.h>
#include <stdint.h>

// ---------------------------------------------------------------------------
// Model dims (fixed by the reference)
// ---------------------------------------------------------------------------
#define LYR   12
#define NHEAD 12
#define HDIM  64
#define HID_  768
#define FF_   3072
#define BB_   2
#define SS_   1024
#define NTOK  (BB_ * SS_)      // 2048

typedef _Float16 f16;
typedef __attribute__((ext_vector_type(16))) _Float16 v16h;
typedef __attribute__((ext_vector_type(8)))  _Float16 v8h;
typedef __attribute__((ext_vector_type(4)))  _Float16 v4h;
typedef __attribute__((ext_vector_type(8)))  float    v8f;

static __device__ __forceinline__ v8f vzero8() {
  v8f z;
#pragma unroll
  for (int i = 0; i < 8; ++i) z[i] = 0.0f;
  return z;
}

static __device__ __forceinline__ v4h pk4(float4 f) {
  v4h h;
  h[0] = (f16)f.x; h[1] = (f16)f.y; h[2] = (f16)f.z; h[3] = (f16)f.w;
  return h;
}

// Assemble a 16-half WMMA fragment from two 16-byte LDS runs.
static __device__ __forceinline__ v16h ld_frag(const f16* lo, const f16* hi) {
  union { v16h v; v8h h[2]; } u;
  u.h[0] = *(const v8h*)lo;
  u.h[1] = *(const v8h*)hi;
  return u.v;
}

static __device__ __forceinline__ v8f wmma16(v16h a, v16h b, v8f c) {
  // (neg_a, A, neg_b, B, c_mod, C, reuse_a, reuse_b)
  return __builtin_amdgcn_wmma_f32_16x16x32_f16(false, a, false, b, (short)0, c,
                                                false, false);
}

// Async global->LDS DMA (CDNA5): 16 bytes per lane, tracked by ASYNCcnt.
static __device__ __forceinline__ void async_ldg_b128(uint32_t lds_off,
                                                      const f16* gptr) {
  asm volatile("global_load_async_to_lds_b128 %0, %1, off"
               :: "v"(lds_off), "v"(gptr)
               : "memory");
}
static __device__ __forceinline__ void wait_asynccnt0() {
  asm volatile("s_wait_asynccnt 0x0" ::: "memory");
}

static __device__ __forceinline__ float gelu_exact(float x) {
  return 0.5f * x * (1.0f + erff(x * 0.70710678118654752f));
}

// ---------------------------------------------------------------------------
// fp32 -> f16 weight conversion (vectorized, n4 = n/4 float4 chunks)
// ---------------------------------------------------------------------------
__global__ __launch_bounds__(256) void cvt_f32_f16_kernel(
    const float* __restrict__ src, f16* __restrict__ dst, int n4) {
  const int i = blockIdx.x * 256 + threadIdx.x;
  if (i >= n4) return;
  const float4 f = ((const float4*)src)[i];
  ((v4h*)dst)[i] = pk4(f);
}

// ---------------------------------------------------------------------------
// Embedding gather + LayerNorm.  One block (256 thr) per token; 768 = 3*256.
// ---------------------------------------------------------------------------
__global__ __launch_bounds__(256) void embed_ln_kernel(
    const int* __restrict__ input_ids, const int* __restrict__ token_type_ids,
    const float* __restrict__ word_emb, const float* __restrict__ tok_emb,
    const float* __restrict__ gw, const float* __restrict__ gb,
    float* __restrict__ out) {
  __shared__ float red[256];
  const int t   = blockIdx.x;
  const int tid = threadIdx.x;
  const size_t wo = (size_t)input_ids[t] * HID_;
  const size_t to = (size_t)token_type_ids[t] * HID_;

  float v0 = word_emb[wo + tid]       + tok_emb[to + tid];
  float v1 = word_emb[wo + tid + 256] + tok_emb[to + tid + 256];
  float v2 = word_emb[wo + tid + 512] + tok_emb[to + tid + 512];

  red[tid] = v0 + v1 + v2;
  __syncthreads();
#pragma unroll
  for (int o = 128; o > 0; o >>= 1) {
    if (tid < o) red[tid] += red[tid + o];
    __syncthreads();
  }
  const float mean = red[0] * (1.0f / HID_);
  __syncthreads();

  float d0 = v0 - mean, d1 = v1 - mean, d2 = v2 - mean;
  red[tid] = d0 * d0 + d1 * d1 + d2 * d2;
  __syncthreads();
#pragma unroll
  for (int o = 128; o > 0; o >>= 1) {
    if (tid < o) red[tid] += red[tid + o];
    __syncthreads();
  }
  const float rinv = rsqrtf(red[0] * (1.0f / HID_) + 1e-12f);

  float* op = out + (size_t)t * HID_;
  op[tid]       = d0 * rinv * gw[tid]       + gb[tid];
  op[tid + 256] = d1 * rinv * gw[tid + 256] + gb[tid + 256];
  op[tid + 512] = d2 * rinv * gw[tid + 512] + gb[tid + 512];
}

// ---------------------------------------------------------------------------
// LayerNorm over last dim (768).  One block per token.
// ---------------------------------------------------------------------------
__global__ __launch_bounds__(256) void layernorm_kernel(
    const float* __restrict__ x, const float* __restrict__ gw,
    const float* __restrict__ gb, float* __restrict__ out) {
  __shared__ float red[256];
  const int t   = blockIdx.x;
  const int tid = threadIdx.x;
  const float* xp = x + (size_t)t * HID_;

  float v0 = xp[tid], v1 = xp[tid + 256], v2 = xp[tid + 512];
  red[tid] = v0 + v1 + v2;
  __syncthreads();
#pragma unroll
  for (int o = 128; o > 0; o >>= 1) {
    if (tid < o) red[tid] += red[tid + o];
    __syncthreads();
  }
  const float mean = red[0] * (1.0f / HID_);
  __syncthreads();

  float d0 = v0 - mean, d1 = v1 - mean, d2 = v2 - mean;
  red[tid] = d0 * d0 + d1 * d1 + d2 * d2;
  __syncthreads();
#pragma unroll
  for (int o = 128; o > 0; o >>= 1) {
    if (tid < o) red[tid] += red[tid + o];
    __syncthreads();
  }
  const float rinv = rsqrtf(red[0] * (1.0f / HID_) + 1e-12f);

  float* op = out + (size_t)t * HID_;
  op[tid]       = d0 * rinv * gw[tid]       + gb[tid];
  op[tid + 256] = d1 * rinv * gw[tid + 256] + gb[tid + 256];
  op[tid + 512] = d2 * rinv * gw[tid + 512] + gb[tid + 512];
}

// ---------------------------------------------------------------------------
// RoPE, in place over a [NTOK, HID] buffer.  One thread per (token, head, i).
// ---------------------------------------------------------------------------
__global__ __launch_bounds__(256) void rope_kernel(float* __restrict__ x) {
  const int idx = blockIdx.x * 256 + threadIdx.x;
  if (idx >= NTOK * NHEAD * (HDIM / 2)) return;
  const int i  = idx & 31;
  const int hh = (idx >> 5) % NHEAD;
  const int t  = idx / (32 * NHEAD);
  const int s  = t & (SS_ - 1);

  float* p = x + (size_t)t * HID_ + hh * HDIM;
  const float inv = __expf(-(float)i * (9.210340371976184f / 32.0f)); // 10000^(-i/32)
  const float ang = (float)s * inv;
  float sn, cs;
  __sincosf(ang, &sn, &cs);
  const float x1 = p[i], x2 = p[i + 32];
  p[i]      = x1 * cs - x2 * sn;
  p[i + 32] = x1 * sn + x2 * cs;
}

// ---------------------------------------------------------------------------
// WMMA GEMM:  out[M,N] = epi(A[M,K]_f32 @ W[N,K]_f16^T + bias [+ res])
//   mode 0: +bias ; mode 1: +bias then GELU ; mode 2: +bias +res
// Block: 256 thr (8 waves), tile 128x128, BK=32, double-buffered LDS.
// A (activations, fp32) pipelines through registers with on-the-fly f16
// convert; W (pre-converted f16) is DMA'd straight into LDS with
// global_load_async_to_lds_b128 (ASYNCcnt).  One barrier per K-step.
// Wave grid 4(M) x 2(N); each wave owns 32x64 = 2x4 WMMA tiles.
// ---------------------------------------------------------------------------
__global__ __launch_bounds__(256) void gemm_wmma_kernel(
    const float* __restrict__ A, const f16* __restrict__ W,
    const float* __restrict__ bias, const float* __restrict__ res,
    float* __restrict__ out, int M, int N, int K, int mode) {
  __shared__ f16 As[2][128][40];   // [buf][m][k], +8 pad (80B row stride)
  __shared__ f16 Bs[2][128][40];   // [buf][n][k]

  const int tid   = threadIdx.x;
  const int lane  = tid & 31;
  const int wid   = tid >> 5;
  const int l15   = lane & 15;
  const int lhi   = lane >> 4;        // half-wave select
  const int waveM = wid & 3;          // 0..3
  const int waveN = wid >> 2;         // 0..1
  const int bm    = blockIdx.y * 128;
  const int bn    = blockIdx.x * 128;

  v8f acc[2][4];
#pragma unroll
  for (int mi = 0; mi < 2; ++mi)
#pragma unroll
    for (int ni = 0; ni < 4; ++ni) acc[mi][ni] = vzero8();

  const int rowS = tid >> 3;          // 0..31
  const int colS = (tid & 7) * 4;     // 0,4,..28
  const size_t aRow = (size_t)(bm + rowS) * K + colS;

  // B async-DMA chunk mapping: 512 x 16B chunks per tile, 2 per thread.
  const int bRow0 = tid >> 2,          bKo0 = (tid & 3) * 8;        // chunk tid
  const int bRow1 = (tid + 256) >> 2,  bKo1 = ((tid + 256) & 3) * 8;
  const f16* Wr0 = W + (size_t)(bn + bRow0) * K + bKo0;
  const f16* Wr1 = W + (size_t)(bn + bRow1) * K + bKo1;

  float4 ra[4];

  // ---- prologue: tile 0 ---------------------------------------------------
  async_ldg_b128((uint32_t)(uintptr_t)&Bs[0][bRow0][bKo0], Wr0);
  async_ldg_b128((uint32_t)(uintptr_t)&Bs[0][bRow1][bKo1], Wr1);
#pragma unroll
  for (int rr = 0; rr < 4; ++rr)
    ra[rr] = *(const float4*)(&A[aRow + (size_t)rr * 32 * K]);
#pragma unroll
  for (int rr = 0; rr < 4; ++rr)
    *(v4h*)&As[0][rowS + rr * 32][colS] = pk4(ra[rr]);
  wait_asynccnt0();
  __syncthreads();

  const int nK = K >> 5;
  for (int kt = 0; kt < nK; ++kt) {
    const int cur = kt & 1;
    const int nxt = cur ^ 1;

    // Issue next tile's loads first: B via async DMA into the other LDS
    // buffer, A into registers (waits resolve after the WMMA block).
    if (kt + 1 < nK) {
      const size_t k1 = (size_t)(kt + 1) * 32;
      async_ldg_b128((uint32_t)(uintptr_t)&Bs[nxt][bRow0][bKo0], Wr0 + k1);
      async_ldg_b128((uint32_t)(uintptr_t)&Bs[nxt][bRow1][bKo1], Wr1 + k1);
#pragma unroll
      for (int rr = 0; rr < 4; ++rr)
        ra[rr] = *(const float4*)(&A[aRow + (size_t)rr * 32 * K + k1]);
    }

    // ---- WMMA on current buffer ----------------------------------------
    v16h af[2], bfr[4];
#pragma unroll
    for (int mi = 0; mi < 2; ++mi) {
      const int r = waveM * 32 + mi * 16 + l15;
      // A layout: lanes 0-15 hold K 0-7 & 16-23; lanes 16-31 hold K 8-15 & 24-31
      af[mi] = ld_frag(&As[cur][r][lhi * 8], &As[cur][r][lhi * 8 + 16]);
    }
#pragma unroll
    for (int ni = 0; ni < 4; ++ni) {
      const int n = waveN * 64 + ni * 16 + l15;
      // B layout: lanes 0-15 hold K 0-7 & 16-23; lanes 16-31 hold K 8-15 & 24-31
      bfr[ni] = ld_frag(&Bs[cur][n][lhi * 8], &Bs[cur][n][lhi * 8 + 16]);
    }
#pragma unroll
    for (int mi = 0; mi < 2; ++mi)
#pragma unroll
      for (int ni = 0; ni < 4; ++ni)
        acc[mi][ni] = wmma16(af[mi], bfr[ni], acc[mi][ni]);

    // ---- stage next A tile into the other buffer -------------------------
    if (kt + 1 < nK) {
#pragma unroll
      for (int rr = 0; rr < 4; ++rr)
        *(v4h*)&As[nxt][rowS + rr * 32][colS] = pk4(ra[rr]);
    }
    wait_asynccnt0();
    __syncthreads();
  }

  // Epilogue.  C layout: VGPR r -> row M = r + 8*lhi; col N = l15.
#pragma unroll
  for (int ni = 0; ni < 4; ++ni) {
    const int col = bn + waveN * 64 + ni * 16 + l15;
    const float bval = bias[col];
#pragma unroll
    for (int mi = 0; mi < 2; ++mi) {
      const int row0 = bm + waveM * 32 + mi * 16 + lhi * 8;
#pragma unroll
      for (int r = 0; r < 8; ++r) {
        const int row = row0 + r;
        float x = acc[mi][ni][r] + bval;
        if (mode == 1) x = gelu_exact(x);
        else if (mode == 2) x += res[(size_t)row * N + col];
        out[(size_t)row * N + col] = x;
      }
    }
  }
}

// ---------------------------------------------------------------------------
// Flash attention (WMMA).  Grid (S/64, H, B); 128 thr = 4 waves; each wave
// owns a 16-row q strip.  K/V processed in 64-key tiles with online softmax.
// ---------------------------------------------------------------------------
__global__ __launch_bounds__(128) void flash_attn_kernel(
    const float* __restrict__ q, const float* __restrict__ k,
    const float* __restrict__ v, const int* __restrict__ amask,
    float* __restrict__ ctx) {
  __shared__ f16 Qs[64][72];
  __shared__ f16 Ks[64][72];
  __shared__ f16 Vs[64][72];   // transposed: Vs[d][key]
  __shared__ f16 Ps[64][72];
  __shared__ float Ms[64];

  const int qt  = blockIdx.x;
  const int hh  = blockIdx.y;
  const int bb  = blockIdx.z;
  const int tid = threadIdx.x;
  const int lane = tid & 31;
  const int wid  = tid >> 5;     // 0..3
  const int l15  = lane & 15;
  const int lhi  = lane >> 4;

  const float* qp = q + ((size_t)(bb * SS_ + qt * 64)) * HID_ + hh * HDIM;
  for (int i = tid; i < 64 * 16; i += 128) {
    const int r = i >> 4, c4 = (i & 15) * 4;
    const float4 f = *(const float4*)(&qp[(size_t)r * HID_ + c4]);
    *(v4h*)&Qs[r][c4] = pk4(f);
  }

  v8f o[4];
#pragma unroll
  for (int i = 0; i < 4; ++i) o[i] = vzero8();
  float mrow[8], lrow[8];
#pragma unroll
  for (int r = 0; r < 8; ++r) { mrow[r] = -INFINITY; lrow[r] = 0.0f; }

  for (int kt = 0; kt < SS_ / 64; ++kt) {
    __syncthreads();
    const float* kp = k + ((size_t)(bb * SS_ + kt * 64)) * HID_ + hh * HDIM;
    const float* vp = v + ((size_t)(bb * SS_ + kt * 64)) * HID_ + hh * HDIM;
    for (int i = tid; i < 64 * 16; i += 128) {
      const int r = i >> 4, c4 = (i & 15) * 4;
      const float4 fk = *(const float4*)(&kp[(size_t)r * HID_ + c4]);
      *(v4h*)&Ks[r][c4] = pk4(fk);
      const float4 fv = *(const float4*)(&vp[(size_t)r * HID_ + c4]);
      Vs[c4 + 0][r] = (f16)fv.x;    // transpose to [d][key]
      Vs[c4 + 1][r] = (f16)fv.y;
      Vs[c4 + 2][r] = (f16)fv.z;
      Vs[c4 + 3][r] = (f16)fv.w;
    }
    if (tid < 64)
      Ms[tid] = (amask[bb * SS_ + kt * 64 + tid] != 0) ? 0.0f : -10000.0f;
    __syncthreads();

    // ---- scores: 16x64 per wave = q_strip @ K^T -------------------------
    v8f s[4];
#pragma unroll
    for (int ni = 0; ni < 4; ++ni) s[ni] = vzero8();
#pragma unroll
    for (int dc = 0; dc < 2; ++dc) {
      const int qr = wid * 16 + l15;
      const v16h aq = ld_frag(&Qs[qr][dc * 32 + lhi * 8],
                              &Qs[qr][dc * 32 + lhi * 8 + 16]);
#pragma unroll
      for (int ni = 0; ni < 4; ++ni) {
        const int kr = ni * 16 + l15;
        const v16h bk = ld_frag(&Ks[kr][dc * 32 + lhi * 8],
                                &Ks[kr][dc * 32 + lhi * 8 + 16]);
        s[ni] = wmma16(aq, bk, s[ni]);
      }
    }

    // scale + additive mask (col = key = ni*16 + l15)
    float madd[4];
#pragma unroll
    for (int ni = 0; ni < 4; ++ni) madd[ni] = Ms[ni * 16 + l15];
#pragma unroll
    for (int ni = 0; ni < 4; ++ni)
#pragma unroll
      for (int r = 0; r < 8; ++r) s[ni][r] = s[ni][r] * 0.125f + madd[ni];

    // ---- online softmax -------------------------------------------------
    float tmax[8];
#pragma unroll
    for (int r = 0; r < 8; ++r) {
      float m = s[0][r];
#pragma unroll
      for (int ni = 1; ni < 4; ++ni) m = fmaxf(m, s[ni][r]);
      tmax[r] = m;
    }
#pragma unroll
    for (int off = 1; off < 16; off <<= 1)
#pragma unroll
      for (int r = 0; r < 8; ++r)
        tmax[r] = fmaxf(tmax[r], __shfl_xor(tmax[r], off, 32));

    float alpha[8], rsum[8];
#pragma unroll
    for (int r = 0; r < 8; ++r) {
      const float mnew = fmaxf(mrow[r], tmax[r]);
      alpha[r] = __expf(mrow[r] - mnew);
      mrow[r]  = mnew;
      rsum[r]  = 0.0f;
    }
#pragma unroll
    for (int ni = 0; ni < 4; ++ni)
#pragma unroll
      for (int r = 0; r < 8; ++r) {
        const float p = __expf(s[ni][r] - mrow[r]);
        s[ni][r] = p;
        rsum[r] += p;
      }
#pragma unroll
    for (int off = 1; off < 16; off <<= 1)
#pragma unroll
      for (int r = 0; r < 8; ++r) rsum[r] += __shfl_xor(rsum[r], off, 32);
#pragma unroll
    for (int r = 0; r < 8; ++r) lrow[r] = lrow[r] * alpha[r] + rsum[r];

    // rescale running output
#pragma unroll
    for (int di = 0; di < 4; ++di)
#pragma unroll
      for (int r = 0; r < 8; ++r) o[di][r] *= alpha[r];

    // ---- P (C layout) -> LDS -> A-fragment layout -----------------------
#pragma unroll
    for (int ni = 0; ni < 4; ++ni)
#pragma unroll
      for (int r = 0; r < 8; ++r)
        Ps[wid * 16 + r + lhi * 8][ni * 16 + l15] = (f16)s[ni][r];
    __syncthreads();

    // ---- o += P @ V  (V^T staged as Vs[d][key]) -------------------------
#pragma unroll
    for (int dc = 0; dc < 2; ++dc) {
      const int pr = wid * 16 + l15;
      const v16h ap = ld_frag(&Ps[pr][dc * 32 + lhi * 8],
                              &Ps[pr][dc * 32 + lhi * 8 + 16]);
#pragma unroll
      for (int di = 0; di < 4; ++di) {
        const int dr = di * 16 + l15;
        const v16h bv = ld_frag(&Vs[dr][dc * 32 + lhi * 8],
                                &Vs[dr][dc * 32 + lhi * 8 + 16]);
        o[di] = wmma16(ap, bv, o[di]);
      }
    }
  }

  // ---- normalize and write ----------------------------------------------
  float* op = ctx + ((size_t)(bb * SS_ + qt * 64 + wid * 16)) * HID_ + hh * HDIM;
#pragma unroll
  for (int di = 0; di < 4; ++di)
#pragma unroll
    for (int r = 0; r < 8; ++r)
      op[(size_t)(r + lhi * 8) * HID_ + di * 16 + l15] = o[di][r] / lrow[r];
}

// ---------------------------------------------------------------------------
// Host-side orchestration
// ---------------------------------------------------------------------------
extern "C" void kernel_launch(void* const* d_in, const int* in_sizes, int n_in,
                              void* d_out, int out_size, void* d_ws,
                              size_t ws_size, hipStream_t stream) {
  const int*   input_ids = (const int*)d_in[0];
  const int*   tok_type  = (const int*)d_in[1];
  const int*   amask     = (const int*)d_in[2];
  const float* word_emb  = (const float*)d_in[3];
  const float* tok_emb   = (const float*)d_in[4];
  const float* ln_e_w    = (const float*)d_in[5];
  const float* ln_e_b    = (const float*)d_in[6];
  const float* Wq        = (const float*)d_in[7];
  const float* bq        = (const float*)d_in[8];
  const float* Wk        = (const float*)d_in[9];
  const float* bk        = (const float*)d_in[10];
  const float* Wv        = (const float*)d_in[11];
  const float* bv        = (const float*)d_in[12];
  const float* Wo        = (const float*)d_in[13];
  const float* bo        = (const float*)d_in[14];
  const float* ln1_w     = (const float*)d_in[15];
  const float* ln1_b     = (const float*)d_in[16];
  const float* Wi        = (const float*)d_in[17];
  const float* bi        = (const float*)d_in[18];
  const float* Wf        = (const float*)d_in[19];
  const float* bf        = (const float*)d_in[20];
  const float* ln2_w     = (const float*)d_in[21];
  const float* ln2_b     = (const float*)d_in[22];

  // Workspace layout: fp32 activations (~63 MB) + f16 weights (~14.2 MB).
  float* h   = (float*)d_ws;            // [NTOK, HID]
  float* qb  = h   + (size_t)NTOK * HID_;
  float* kb  = qb  + (size_t)NTOK * HID_;
  float* vb  = kb  + (size_t)NTOK * HID_;
  float* cx  = vb  + (size_t)NTOK * HID_;
  float* tmp = cx  + (size_t)NTOK * HID_;
  float* ff  = tmp + (size_t)NTOK * HID_;   // [NTOK, FF]

  f16* wq16 = (f16*)(ff + (size_t)NTOK * FF_);
  f16* wk16 = wq16 + (size_t)HID_ * HID_;
  f16* wv16 = wk16 + (size_t)HID_ * HID_;
  f16* wo16 = wv16 + (size_t)HID_ * HID_;
  f16* wi16 = wo16 + (size_t)HID_ * HID_;
  f16* wf16 = wi16 + (size_t)FF_ * HID_;

  const int nHH4 = HID_ * HID_ / 4;   // float4 chunks per 768x768 matrix
  const int nFH4 = FF_ * HID_ / 4;
  const int cbHH = (nHH4 + 255) / 256;
  const int cbFH = (nFH4 + 255) / 256;

  const dim3 gHid(HID_ / 128, NTOK / 128);  // (6,16)
  const dim3 gFF(FF_ / 128, NTOK / 128);    // (24,16)
  const dim3 gAtt(SS_ / 64, NHEAD, BB_);    // (16,12,2)
  const int  ropeBlocks = (NTOK * NHEAD * (HDIM / 2) + 255) / 256;

  embed_ln_kernel<<<NTOK, 256, 0, stream>>>(input_ids, tok_type, word_emb,
                                            tok_emb, ln_e_w, ln_e_b, h);

  for (int l = 0; l < LYR; ++l) {
    const float* Wq_l = Wq + (size_t)l * HID_ * HID_;
    const float* Wk_l = Wk + (size_t)l * HID_ * HID_;
    const float* Wv_l = Wv + (size_t)l * HID_ * HID_;
    const float* Wo_l = Wo + (size_t)l * HID_ * HID_;
    const float* Wi_l = Wi + (size_t)l * FF_ * HID_;
    const float* Wf_l = Wf + (size_t)l * HID_ * FF_;
    const float* bq_l = bq + (size_t)l * HID_;
    const float* bk_l = bk + (size_t)l * HID_;
    const float* bv_l = bv + (size_t)l * HID_;
    const float* bo_l = bo + (size_t)l * HID_;
    const float* bi_l = bi + (size_t)l * FF_;
    const float* bf_l = bf + (size_t)l * HID_;

    // Convert this layer's weights to f16 once; GEMMs then stream them as
    // half-width async-LDS DMAs.
    cvt_f32_f16_kernel<<<cbHH, 256, 0, stream>>>(Wq_l, wq16, nHH4);
    cvt_f32_f16_kernel<<<cbHH, 256, 0, stream>>>(Wk_l, wk16, nHH4);
    cvt_f32_f16_kernel<<<cbHH, 256, 0, stream>>>(Wv_l, wv16, nHH4);
    cvt_f32_f16_kernel<<<cbHH, 256, 0, stream>>>(Wo_l, wo16, nHH4);
    cvt_f32_f16_kernel<<<cbFH, 256, 0, stream>>>(Wi_l, wi16, nFH4);
    cvt_f32_f16_kernel<<<cbFH, 256, 0, stream>>>(Wf_l, wf16, nFH4);

    gemm_wmma_kernel<<<gHid, 256, 0, stream>>>(h, wq16, bq_l, nullptr, qb,
                                               NTOK, HID_, HID_, 0);
    gemm_wmma_kernel<<<gHid, 256, 0, stream>>>(h, wk16, bk_l, nullptr, kb,
                                               NTOK, HID_, HID_, 0);
    gemm_wmma_kernel<<<gHid, 256, 0, stream>>>(h, wv16, bv_l, nullptr, vb,
                                               NTOK, HID_, HID_, 0);
    rope_kernel<<<ropeBlocks, 256, 0, stream>>>(qb);
    rope_kernel<<<ropeBlocks, 256, 0, stream>>>(kb);

    flash_attn_kernel<<<gAtt, 128, 0, stream>>>(qb, kb, vb, amask, cx);

    // attn output projection + residual, then LN1 -> h
    gemm_wmma_kernel<<<gHid, 256, 0, stream>>>(cx, wo16, bo_l, h, tmp,
                                               NTOK, HID_, HID_, 2);
    layernorm_kernel<<<NTOK, 256, 0, stream>>>(tmp,
                                               ln1_w + (size_t)l * HID_,
                                               ln1_b + (size_t)l * HID_, h);

    // FFN: gelu(h @ Wi^T + bi) @ Wf^T + bf + h, then LN2
    gemm_wmma_kernel<<<gFF, 256, 0, stream>>>(h, wi16, bi_l, nullptr, ff,
                                              NTOK, FF_, HID_, 1);
    gemm_wmma_kernel<<<gHid, 256, 0, stream>>>(ff, wf16, bf_l, h, tmp,
                                               NTOK, HID_, FF_, 2);
    float* lnout = (l == LYR - 1) ? (float*)d_out : h;
    layernorm_kernel<<<NTOK, 256, 0, stream>>>(tmp,
                                               ln2_w + (size_t)l * HID_,
                                               ln2_b + (size_t)l * HID_, lnout);
  }
}
